// CausalMultiHeadAttention_66520453480513
// MI455X (gfx1250) — matrix operand order repaired and checked by reading, not compiled, not run on hardware
//
#include <hip/hip_runtime.h>

typedef __attribute__((ext_vector_type(16))) _Float16 v16h;
typedef __attribute__((ext_vector_type(8)))  _Float16 v8h;
typedef __attribute__((ext_vector_type(8)))  float    v8f;
typedef __attribute__((ext_vector_type(4)))  unsigned int u32x4;
typedef __attribute__((ext_vector_type(8)))  int          i32x8;
typedef __attribute__((ext_vector_type(4)))  int          i32x4;

static __device__ __forceinline__ v8f wmma16x16x32(v16h a, v16h b, v8f c) {
  // D = A(16x32 f16) * B(32x16 f16) + C(16x16 f32)
  return __builtin_amdgcn_wmma_f32_16x16x32_f16(false, a, false, b, (short)0, c, false, false);
}

static __device__ __forceinline__ v16h cat16(v8h lo, v8h hi) {
  return __builtin_shufflevector(lo, hi, 0,1,2,3,4,5,6,7,8,9,10,11,12,13,14,15);
}

// A fragment (16x32 f16) from a row-major [16][stride] tile.
// ISA 7.12.2: lane holds row = lane&15; v0..3 = K (lane>>4)*8+0..7, v4..7 = 16+(lane>>4)*8+0..7
static __device__ __forceinline__ v16h load_frag_a(const _Float16* base, int stride, int lane) {
  const _Float16* p = base + (lane & 15) * stride + ((lane >> 4) << 3);
  return cat16(*(const v8h*)p, *(const v8h*)(p + 16));
}

// B fragment (32x16 f16) from a transposed [n][k] tile with row stride `stride`.
// lane holds column n = lane&15; v0..3 = K (lane>>4)*8+0..7, v4..7 = 16+(lane>>4)*8+0..7
static __device__ __forceinline__ v16h load_frag_b(const _Float16* base, int stride, int lane) {
  const _Float16* p = base + (lane & 15) * stride + ((lane >> 4) << 3);
  return cat16(*(const v8h*)p, *(const v8h*)(p + 16));
}

// reductions within a 16-lane half of a wave32 (C/D layout keeps a row inside one half)
static __device__ __forceinline__ float hmax16(float v) {
#pragma unroll
  for (int m = 1; m <= 8; m <<= 1) v = fmaxf(v, __shfl_xor(v, m, 32));
  return v;
}
static __device__ __forceinline__ float hsum16(float v) {
#pragma unroll
  for (int m = 1; m <= 8; m <<= 1) v += __shfl_xor(v, m, 32);
  return v;
}

// ---------------------------------------------------------------------------
// TDM: issue a 2-D f16 tile load (tile0 x tile1 elements, global row stride
// stride0 elements) into LDS, with 16B padding after every 128B row so the
// LDS image has row stride 72 halfwords.  D# per ISA ch.8.  TENSORcnt-tracked.
// ---------------------------------------------------------------------------
static __device__ __forceinline__ void tdm_load_2d_f16(
    unsigned lds_addr, unsigned long long gaddr,
    unsigned tile0, unsigned tile1, unsigned long long stride0)
{
  u32x4 g0;
  g0[0] = 1u;                                             // count=1, no gather
  g0[1] = lds_addr;                                       // LDS byte address
  g0[2] = (unsigned)(gaddr & 0xffffffffull);              // global_addr[31:0]
  g0[3] = (unsigned)((gaddr >> 32) & 0x1ffffffull)        // global_addr[56:32]
        | (2u << 30);                                     // type = 2 ("image")
  i32x8 g1;
  g1[0] = (int)((1u << 16)      // data_size = 2 bytes
              | (1u << 20)      // pad_enable
              | (4u << 22)      // pad_interval: 32 DWORDs (=128B row)
              | (3u << 25));    // pad_amount:   4 DWORDs (=16B -> stride 144B)
  g1[1] = (int)(tile0 << 16);                 // tensor_dim0[15:0]
  g1[2] = (int)(tile1 << 16);                 // tensor_dim0 hi=0 | tensor_dim1 lo
  g1[3] = (int)(tile0 << 16);                 // tensor_dim1 hi=0 | tile_dim0
  g1[4] = (int)tile1;                         // tile_dim1 | tile_dim2=0
  g1[5] = (int)(stride0 & 0xffffffffull);     // tensor_dim0_stride[31:0]
  g1[6] = (int)((stride0 >> 32) & 0xffffull); // stride0[47:32] | stride1 lo = 0
  g1[7] = 0;
  const i32x4 z4 = {0, 0, 0, 0};              // groups 2/3 unused (2-D tensor)
  const i32x8 z8 = {0, 0, 0, 0, 0, 0, 0, 0};  // trailing group (clang-23 form)
  __builtin_amdgcn_tensor_load_to_lds(g0, g1, z4, z4, z8, 0);
}

// ---------------------------------------------------------------------------
// Kernel 1: fused QKV projection.  C = X[8192x1024] * W[1024x1024] (3 mats).
// Output: Q,K as f16 [B,H,S,64]; V transposed as f16 [B,H,64,S].
// ---------------------------------------------------------------------------
__global__ __launch_bounds__(256) void qkv_proj_kernel(
    const float* __restrict__ X,
    const float* __restrict__ Wq, const float* __restrict__ Wk, const float* __restrict__ Wv,
    _Float16* __restrict__ Qo, _Float16* __restrict__ Ko, _Float16* __restrict__ Vt)
{
  constexpr int LD = 40;                       // padded (32 + 8) to dodge bank conflicts
  __shared__ _Float16 As[2][128 * LD];         // X tile, [m][k]
  __shared__ _Float16 Bs[2][128 * LD];         // W tile, transposed [n][k]

  const int tid  = threadIdx.x;
  const int lane = tid & 31;
  const int w    = tid >> 5;
  const int wm   = w >> 2;                     // 0..1  (64 rows each)
  const int wn   = w & 3;                      // 0..3  (32 cols each)
  const int m0   = blockIdx.y * 128;
  const int n0g  = blockIdx.x * 128;           // 0..3071 across Wq|Wk|Wv
  const int mat  = n0g >> 10;
  const int col0 = n0g & 1023;
  const float* __restrict__ W = (mat == 0) ? Wq : (mat == 1) ? Wk : Wv;

  const int xr = tid >> 2;                     // X: row pair base
  const int xc = (tid & 3) << 3;               // X: 8-col group
  const int wk = tid & 31;                     // W: K row
  const int wc = (tid >> 5) << 4;              // W: 16-col group

  auto load_tile = [&](int kt, int buf) {
    const int k0 = kt << 5;
#pragma unroll
    for (int rr = 0; rr < 2; ++rr) {
      const int row = xr + rr * 64;
      const float4* gp = (const float4*)(X + (size_t)(m0 + row) * 1024 + k0 + xc);
      const float4 aa = gp[0], bb = gp[1];
      _Float16* dst = &As[buf][row * LD + xc];
      dst[0] = (_Float16)aa.x; dst[1] = (_Float16)aa.y;
      dst[2] = (_Float16)aa.z; dst[3] = (_Float16)aa.w;
      dst[4] = (_Float16)bb.x; dst[5] = (_Float16)bb.y;
      dst[6] = (_Float16)bb.z; dst[7] = (_Float16)bb.w;
    }
    const float4* wp = (const float4*)(W + (size_t)(k0 + wk) * 1024 + col0 + wc);
    const float4 c0 = wp[0], c1 = wp[1], c2 = wp[2], c3 = wp[3];
    const float vals[16] = {c0.x,c0.y,c0.z,c0.w, c1.x,c1.y,c1.z,c1.w,
                            c2.x,c2.y,c2.z,c2.w, c3.x,c3.y,c3.z,c3.w};
#pragma unroll
    for (int i = 0; i < 16; ++i)
      Bs[buf][(wc + i) * LD + wk] = (_Float16)vals[i];   // transpose into [n][k]
  };

  v8f acc[4][2] = {};

  load_tile(0, 0);
  __syncthreads();
  for (int kt = 0; kt < 32; ++kt) {
    const int buf = kt & 1;
    if (kt + 1 < 32) load_tile(kt + 1, buf ^ 1);         // double-buffered prefetch
    v16h af[4], bf[2];
#pragma unroll
    for (int mi = 0; mi < 4; ++mi)
      af[mi] = load_frag_a(&As[buf][(wm * 64 + mi * 16) * LD], LD, lane);
#pragma unroll
    for (int ni = 0; ni < 2; ++ni)
      bf[ni] = load_frag_b(&Bs[buf][(wn * 32 + ni * 16) * LD], LD, lane);
#pragma unroll
    for (int mi = 0; mi < 4; ++mi)
#pragma unroll
      for (int ni = 0; ni < 2; ++ni)
        acc[mi][ni] = wmma16x16x32(af[mi], bf[ni], acc[mi][ni]);
    __syncthreads();
  }

  const int hh = lane >> 4, ln = lane & 15;
#pragma unroll
  for (int mi = 0; mi < 4; ++mi) {
#pragma unroll
    for (int ni = 0; ni < 2; ++ni) {
#pragma unroll
      for (int r = 0; r < 8; ++r) {
        const int grow = m0 + wm * 64 + mi * 16 + r + 8 * hh;     // C/D layout row
        const int gcol = col0 + wn * 32 + ni * 16 + ln;           // C/D layout col
        const int b = grow >> 11, s = grow & 2047;
        const int h = gcol >> 6,  d = gcol & 63;
        const _Float16 hv = (_Float16)acc[mi][ni][r];
        if (mat == 0)      Qo[((size_t)((b * 16 + h) * 2048 + s) << 6) + d] = hv;
        else if (mat == 1) Ko[((size_t)((b * 16 + h) * 2048 + s) << 6) + d] = hv;
        else               Vt[((size_t)((b * 16 + h) * 64 + d) << 11) + s] = hv;
      }
    }
  }
}

// ---------------------------------------------------------------------------
// Kernel 2: causal flash attention.  One WG = 128 queries of one (b,h);
// 8 waves x 16 query rows; key chunks of 64 double-buffered via TDM.
// ---------------------------------------------------------------------------
__global__ __launch_bounds__(256) void attn_kernel(
    const _Float16* __restrict__ Qm, const _Float16* __restrict__ Km,
    const _Float16* __restrict__ Vt, float* __restrict__ Out)
{
  constexpr int LDK = 72;                      // 64 + 8 (TDM pads 16B per 128B row)
  __shared__ _Float16 Ks[2][64 * LDK];         // [key][d]
  __shared__ _Float16 Vs[2][64 * LDK];         // [d][key]  (from transposed V)
  __shared__ _Float16 Ps[8 * 16 * LDK];        // per-wave P transpose scratch

  const int tid  = threadIdx.x;
  const int lane = tid & 31;
  const int w    = tid >> 5;
  const int hh   = lane >> 4;
  const int ln   = lane & 15;
  const int bh   = blockIdx.y;                 // b*16 + h
  const int b    = bh >> 4;
  const int h    = bh & 15;
  const int q0   = blockIdx.x * 128;
  const int qw0  = q0 + w * 16;                // this wave's first query row

  constexpr float CEXP = 1.4426950408889634f / 32.0f;   // log2(e) / sqrt(QK_DIM)

  const int nch = blockIdx.x * 2 + 2;          // causal: chunks 0 .. (q0+128)/64 - 1

  // wave 0 drives the Tensor Data Mover (EXEC-independent, wave-uniform branch)
  auto issue_chunk = [&](int ci, int buf) {
    const int kc = ci << 6;
    tdm_load_2d_f16((unsigned)(size_t)&Ks[buf][0],
                    (unsigned long long)(size_t)(Km + ((size_t)(bh * 2048 + kc) << 6)),
                    64, 64, 64);
    tdm_load_2d_f16((unsigned)(size_t)&Vs[buf][0],
                    (unsigned long long)(size_t)(Vt + (((size_t)bh << 6) << 11) + kc),
                    64, 64, 2048);
  };

  if (w == 0) issue_chunk(0, 0);

  // Q fragments held in registers for the whole kernel (d = 2 chunks of 32)
  const _Float16* qbase = Qm + ((size_t)(bh * 2048 + qw0) << 6);
  v16h qf[2];
#pragma unroll
  for (int c = 0; c < 2; ++c) qf[c] = load_frag_a(qbase + c * 32, 64, lane);

  v8f o[4] = {};
  float m_i[8], l_i[8];
#pragma unroll
  for (int r = 0; r < 8; ++r) { m_i[r] = -1e30f; l_i[r] = 0.0f; }

  for (int ci = 0; ci < nch; ++ci) {
    const int buf = ci & 1;
    if (w == 0) __builtin_amdgcn_s_wait_tensorcnt(0);   // chunk ci landed in LDS
    __syncthreads();                                    // publish to all waves;
                                                        // also: everyone is done
                                                        // reading buf^1 (chunk ci-1)
    if (w == 0 && ci + 1 < nch) issue_chunk(ci + 1, buf ^ 1);   // async prefetch

    const int kc = ci << 6;
    const _Float16* ks = &Ks[buf][0];
    const _Float16* vs = &Vs[buf][0];

    // scores: 16 queries x 64 keys (4 column tiles), K-dim = 64 (2 WMMAs)
    v8f sc[4];
#pragma unroll
    for (int tn = 0; tn < 4; ++tn) {
      v8f z = {};
      z = wmma16x16x32(qf[0], load_frag_b(ks + (tn * 16) * LDK,      LDK, lane), z);
      z = wmma16x16x32(qf[1], load_frag_b(ks + (tn * 16) * LDK + 32, LDK, lane), z);
      sc[tn] = z;
    }

    if (kc + 63 > qw0) {                       // chunk touches/crosses the diagonal
#pragma unroll
      for (int tn = 0; tn < 4; ++tn)
#pragma unroll
        for (int r = 0; r < 8; ++r) {
          const int col = kc + tn * 16 + ln;
          const int row = qw0 + r + 8 * hh;
          if (col > row) sc[tn][r] = -1e30f;
        }
    }

    // online softmax per row (row r+8*hh lives across the 16 lanes of this half)
#pragma unroll
    for (int r = 0; r < 8; ++r) {
      float mx = fmaxf(fmaxf(sc[0][r], sc[1][r]), fmaxf(sc[2][r], sc[3][r]));
      mx = hmax16(mx);
      const float mnew  = fmaxf(m_i[r], mx);
      const float alpha = __builtin_exp2f((m_i[r] - mnew) * CEXP);
      m_i[r] = mnew;
      float rs = 0.0f;
#pragma unroll
      for (int tn = 0; tn < 4; ++tn) {
        const float p = __builtin_exp2f((sc[tn][r] - mnew) * CEXP);
        sc[tn][r] = p;
        rs += p;
      }
      rs = hsum16(rs);
      l_i[r] = l_i[r] * alpha + rs;
#pragma unroll
      for (int tn = 0; tn < 4; ++tn) o[tn][r] *= alpha;
    }

    // C/D layout -> A layout via per-wave LDS bounce (DS ops in-order per wave)
    _Float16* pw = &Ps[w * 16 * LDK];
#pragma unroll
    for (int tn = 0; tn < 4; ++tn)
#pragma unroll
      for (int r = 0; r < 8; ++r)
        pw[(r + 8 * hh) * LDK + tn * 16 + ln] = (_Float16)sc[tn][r];

    v16h pf[2];
#pragma unroll
    for (int c = 0; c < 2; ++c) pf[c] = load_frag_a(pw + c * 32, LDK, lane);

    // O += P * V  (d = 4 column tiles, keys = 2 chunks of 32)
#pragma unroll
    for (int tn = 0; tn < 4; ++tn) {
      o[tn] = wmma16x16x32(pf[0], load_frag_b(vs + (tn * 16) * LDK,      LDK, lane), o[tn]);
      o[tn] = wmma16x16x32(pf[1], load_frag_b(vs + (tn * 16) * LDK + 32, LDK, lane), o[tn]);
    }
  }

  // normalize and write out (f32, layout [B,S,H*64])
#pragma unroll
  for (int r = 0; r < 8; ++r) {
    const float inv = 1.0f / l_i[r];
    const int row = qw0 + r + 8 * hh;
#pragma unroll
    for (int tn = 0; tn < 4; ++tn) {
      const int d = tn * 16 + ln;
      Out[((size_t)(b * 2048 + row) << 10) + h * 64 + d] = o[tn][r] * inv;
    }
  }
}

extern "C" void kernel_launch(void* const* d_in, const int* in_sizes, int n_in,
                              void* d_out, int out_size, void* d_ws, size_t ws_size,
                              hipStream_t stream) {
  (void)in_sizes; (void)n_in; (void)out_size; (void)ws_size;
  const float* X  = (const float*)d_in[0];
  const float* Wq = (const float*)d_in[1];
  const float* Wk = (const float*)d_in[2];
  const float* Wv = (const float*)d_in[3];
  float* Out = (float*)d_out;

  // workspace: Q,K f16 [B,H,S,64]; V transposed f16 [B,H,64,S]  (48 MB total)
  _Float16* Qb = (_Float16*)d_ws;
  _Float16* Kb = Qb + (size_t)8192 * 1024;
  _Float16* Vt = Kb + (size_t)8192 * 1024;

  qkv_proj_kernel<<<dim3(24, 64), 256, 0, stream>>>(X, Wq, Wk, Wv, Qb, Kb, Vt);
  attn_kernel<<<dim3(16, 64), 256, 0, stream>>>(Qb, Kb, Vt, Out);
}
